// LinearIDOL_29592324669541
// MI455X (gfx1250) — matrix operand order repaired and checked by reading, not compiled, not run on hardware
//
#include <hip/hip_runtime.h>
#include <hip/hip_bf16.h>

// ---------------------------------------------------------------------------
// LinearIDOL on MI455X (gfx1250, wave32, WMMA).
// All big matmuls run through v_wmma_f32_16x16x32_bf16 with f32 accumulation;
// fp32 operands are converted to bf16 (RNE) while staging tiles into LDS with
// vectorized (b128) loads.  Sum|Bs| is fused into the GEMM staging pass so Bs
// (1.34 GB) crosses HBM exactly once: ~58 us @ 23.3 TB/s lower bound.
// ---------------------------------------------------------------------------

typedef __attribute__((ext_vector_type(16))) __bf16        v16bf;
typedef __attribute__((ext_vector_type(8)))  float         v8f;
typedef __attribute__((ext_vector_type(4)))  float         v4f;
typedef __attribute__((ext_vector_type(4)))  unsigned int  v4u;
typedef __attribute__((ext_vector_type(2)))  unsigned int  v2u;

static constexpr int Bsz = 512;
static constexpr int Dd  = 1024;
static constexpr int Hh  = 4096;
static constexpr int TAU = 20;
static constexpr int Tt  = 21;          // TAU + 1
static constexpr int KTOP = 100;
static constexpr int BK  = 32;          // K-slab per LDS stage (one WMMA K)

__device__ __forceinline__ unsigned short f2bf(float f) {
  unsigned u = __builtin_bit_cast(unsigned, f);
  u += 0x7FFFu + ((u >> 16) & 1u);      // round-to-nearest-even
  return (unsigned short)(u >> 16);
}
__device__ __forceinline__ unsigned pk2bf(float a, float b) {
  return (unsigned)f2bf(a) | ((unsigned)f2bf(b) << 16);
}

union FragB { v16bf v; v4u q[2]; };     // 16 bf16 = 8 VGPRs
union Acc8  { v8f  v; float f[8]; };    // 16x16 f32 C/D tile slice per lane

// ---------------------------------------------------------------------------
// Generic WMMA GEMM.  MODE 0: Zp[t] = Xp[:,:,t] x F_enc        (K=1024)
//                     MODE 1: Zt    = sum_l Zp[tau-1-l] x Bs[l]^T
//                                     + Zp[tau] x tril(M,1)^T  (K=4096 each)
//                             (+ fused per-block partial of sum|Bs|)
//                     MODE 2: fused mean((Zlast x F_dec) - Xlast)^2 partials
// LDS layout: A as [m][k], B as [n][k] (bf16, +8 ushort row pad), matching the
// gfx1250 WMMA operand striping:
//   A: lane<16 -> row m, K {0..7,16..23}; lane>=16 -> K {8..15,24..31}
//   B: lane<16 -> col n, K {0..7,16..23}; lane>=16 -> K {8..15,24..31}
//   D: VGPR r  -> (M = r + 8*(lane>=16), N = lane%16)
// ---------------------------------------------------------------------------
template<int MODE, int BM, int BN, int WR, int WC>
__global__ __launch_bounds__(256)
void gemm_kernel(const float* __restrict__ Xp,
                 const float* __restrict__ Fenc,
                 const float* __restrict__ BsT,
                 const float* __restrict__ Mm,
                 const float* __restrict__ Fdec,
                 unsigned short* __restrict__ Zp16,
                 float* __restrict__ Zlast,
                 float* __restrict__ Zt,
                 float* __restrict__ part)
{
  constexpr int LDA  = BK + 8;                 // pad to spread LDS banks
  constexpr int MSUB = BM / (16 * WR);
  constexpr int NSUB = BN / (16 * WC);

  __shared__ __align__(16) unsigned short sA[BM * LDA];
  __shared__ __align__(16) unsigned short sB[BN * LDA];

  const int tid  = threadIdx.x;
  const int lane = tid & 31, wave = tid >> 5;
  const int wm   = wave % WR, wn = wave / WR;
  const int l16  = lane & 15, hi = lane >> 4;
  const int m0   = blockIdx.y * BM;
  const int n0   = blockIdx.x * BN;
  const int t    = (MODE == 0) ? (int)blockIdx.z : 0;

  Acc8 acc[MSUB][NSUB];
  #pragma unroll
  for (int i = 0; i < MSUB; ++i)
    #pragma unroll
    for (int j = 0; j < NSUB; ++j)
      #pragma unroll
      for (int r = 0; r < 8; ++r) acc[i][j].f[r] = 0.0f;

  float babs = 0.0f;                           // fused sum|Bs| (MODE 1 only)
  const int nseg = (MODE == 1) ? (TAU + 1) : 1;
  const int K    = (MODE == 0) ? Dd : Hh;

  for (int seg = 0; seg < nseg; ++seg) {
    for (int kb = 0; kb < K; kb += BK) {
      __syncthreads();
      // ---- stage A tile (BM x BK) into LDS as bf16 ----
      if constexpr (MODE == 0) {
        // Xp slice: inherently strided (elem stride = Tt floats) -> scalar
        for (int e = tid; e < BM * BK; e += 256) {
          int ml = e >> 5, kl = e & 31;
          float v = Xp[(size_t)(m0 + ml) * (Dd * Tt) + (size_t)(kb + kl) * Tt + t];
          sA[ml * LDA + kl] = f2bf(v);
        }
      } else {
        // bf16 panel from workspace: 16-byte vector copies
        const int slice = (MODE == 1) ? (seg < TAU ? (TAU - 1 - seg) : TAU) : TAU;
        const unsigned short* Ap = Zp16 + (size_t)slice * Bsz * Hh;
        #pragma unroll
        for (int i = 0; i < (BM * BK / 8) / 256; ++i) {
          int v  = tid + i * 256;
          int ml = v >> 2, kc = v & 3;         // 4 x 8-ushort chunks per row
          v4u x = *(const v4u*)&Ap[(size_t)(m0 + ml) * Hh + kb + kc * 8];
          *(v4u*)&sA[ml * LDA + kc * 8] = x;
        }
      }
      // ---- stage B tile into LDS as bf16, layout [n][k] ----
      if constexpr (MODE == 1) {               // B is (N,K) row-major: Bs[l] or M
        const bool msk = (seg == TAU);
        const float* Bp = msk ? Mm : (BsT + (size_t)seg * Hh * Hh);
        #pragma unroll
        for (int i = 0; i < (BN * BK / 4) / 256; ++i) {
          int v  = tid + i * 256;
          int nl = v >> 3, kv = v & 7;         // 8 float4 per 32-wide row
          int gn = n0 + nl, gk = kb + kv * 4;
          v4f x = *(const v4f*)&Bp[(size_t)gn * Hh + gk];
          if (msk) {
            #pragma unroll
            for (int c = 0; c < 4; ++c)
              if (gk + c > gn + 1) x[c] = 0.0f;     // tril(M, k=1)
          } else {
            babs += fabsf(x[0]) + fabsf(x[1]) + fabsf(x[2]) + fabsf(x[3]);
          }
          v2u p; p[0] = pk2bf(x[0], x[1]); p[1] = pk2bf(x[2], x[3]);
          *(v2u*)&sB[nl * LDA + kv * 4] = p;
        }
      } else {                                 // B is (K,N) row-major: F_enc/F_dec
        const float* Bp  = (MODE == 0) ? Fenc : Fdec;
        const int Nstr   = (MODE == 0) ? Hh : Dd;
        #pragma unroll
        for (int i = 0; i < (BN * BK / 4) / 256; ++i) {
          int v  = tid + i * 256;
          int kl = v >> 5, nv = v & 31;        // 32 float4 per 128-wide row
          v4f x = *(const v4f*)&Bp[(size_t)(kb + kl) * Nstr + n0 + nv * 4];
          sB[(nv * 4 + 0) * LDA + kl] = f2bf(x[0]);
          sB[(nv * 4 + 1) * LDA + kl] = f2bf(x[1]);
          sB[(nv * 4 + 2) * LDA + kl] = f2bf(x[2]);
          sB[(nv * 4 + 3) * LDA + kl] = f2bf(x[3]);
        }
      }
      __syncthreads();

      // ---- load fragments & issue WMMAs ----
      FragB af[MSUB], bfr[NSUB];
      #pragma unroll
      for (int i = 0; i < MSUB; ++i) {
        int row = wm * (MSUB * 16) + i * 16 + l16;
        int ka  = hi * 8;
        af[i].q[0] = *(const v4u*)&sA[row * LDA + ka];
        af[i].q[1] = *(const v4u*)&sA[row * LDA + ka + 16];
      }
      #pragma unroll
      for (int j = 0; j < NSUB; ++j) {
        int col = wn * (NSUB * 16) + j * 16 + l16;
        int kB  = hi * 8;
        bfr[j].q[0] = *(const v4u*)&sB[col * LDA + kB];
        bfr[j].q[1] = *(const v4u*)&sB[col * LDA + kB + 16];
      }
      #pragma unroll
      for (int i = 0; i < MSUB; ++i)
        #pragma unroll
        for (int j = 0; j < NSUB; ++j)
          acc[i][j].v = __builtin_amdgcn_wmma_f32_16x16x32_bf16(
              false, af[i].v, false, bfr[j].v, (short)0, acc[i][j].v, false, false);
    }
  }

  // ---- epilogue ----
  if constexpr (MODE == 0) {
    #pragma unroll
    for (int i = 0; i < MSUB; ++i)
      #pragma unroll
      for (int j = 0; j < NSUB; ++j) {
        int rg = m0 + wm * (MSUB * 16) + i * 16 + hi * 8;
        int cg = n0 + wn * (NSUB * 16) + j * 16 + l16;
        #pragma unroll
        for (int r = 0; r < 8; ++r) {
          size_t o = (size_t)(rg + r) * Hh + cg;
          Zp16[(size_t)t * Bsz * Hh + o] = f2bf(acc[i][j].f[r]);
          if (t == TAU) Zlast[o] = acc[i][j].f[r];
        }
      }
  } else if constexpr (MODE == 1) {
    #pragma unroll
    for (int i = 0; i < MSUB; ++i)
      #pragma unroll
      for (int j = 0; j < NSUB; ++j) {
        int rg = m0 + wm * (MSUB * 16) + i * 16 + hi * 8;
        int cg = n0 + wn * (NSUB * 16) + j * 16 + l16;
        #pragma unroll
        for (int r = 0; r < 8; ++r)
          Zt[(size_t)(rg + r) * Hh + cg] = acc[i][j].f[r];
      }
    // fused sum|Bs| partial (each Bs element staged once per M-tile; the
    // finalize kernel divides by gridDim.y = Bsz/BM to undo the multiplicity)
    __shared__ float redb[256];
    redb[tid] = babs; __syncthreads();
    for (int off = 128; off; off >>= 1) {
      if (tid < off) redb[tid] += redb[tid + off];
      __syncthreads();
    }
    if (tid == 0) part[blockIdx.y * gridDim.x + blockIdx.x] = redb[0];
  } else {  // MODE == 2: fused (recons - Xp[:,:,-1])^2 partial sum
    float s = 0.0f;
    #pragma unroll
    for (int i = 0; i < MSUB; ++i)
      #pragma unroll
      for (int j = 0; j < NSUB; ++j) {
        int rg = m0 + wm * (MSUB * 16) + i * 16 + hi * 8;
        int cg = n0 + wn * (NSUB * 16) + j * 16 + l16;
        #pragma unroll
        for (int r = 0; r < 8; ++r) {
          float x = Xp[(size_t)(rg + r) * (Dd * Tt) + (size_t)cg * Tt + TAU];
          float d = acc[i][j].f[r] - x;
          s += d * d;
        }
      }
    __shared__ float red[256];
    red[tid] = s; __syncthreads();
    for (int off = 128; off; off >>= 1) {
      if (tid < off) red[tid] += red[tid + off];
      __syncthreads();
    }
    if (tid == 0) part[blockIdx.y * gridDim.x + blockIdx.x] = red[0];
  }
}

// ---------------------------------------------------------------------------
// Per-row top-100 of |Zt| via 4-pass radix select on float bit patterns
// (positive-float bit order == value order), deterministic tie-break by index,
// plus the three masked-Zt loss partial sums.
// ---------------------------------------------------------------------------
__global__ __launch_bounds__(256)
void topk_kernel(const float* __restrict__ Zt, const float* __restrict__ Zl,
                 float* __restrict__ part3)
{
  __shared__ unsigned su[Hh];        // 16 KB: |z| bit patterns
  __shared__ unsigned hist[256];
  __shared__ unsigned s_pref;
  __shared__ int      s_k;
  const int b = blockIdx.x, t = threadIdx.x;
  const float* zr = Zt + (size_t)b * Hh;

  for (int i = t; i < Hh; i += 256)
    su[i] = __builtin_bit_cast(unsigned, fabsf(zr[i]));
  if (t == 0) { s_pref = 0u; s_k = KTOP; }
  __syncthreads();

  for (int pass = 0; pass < 4; ++pass) {
    const int sh = 24 - pass * 8;
    hist[t] = 0;
    __syncthreads();
    const unsigned pref  = s_pref;
    const unsigned hmask = (pass == 0) ? 0u : (0xFFFFFFFFu << (sh + 8));
    for (int i = t; i < Hh; i += 256) {
      unsigned u = su[i];
      if ((u & hmask) == pref) atomicAdd(&hist[(u >> sh) & 255u], 1u);
    }
    __syncthreads();
    if (t == 0) {
      int kk = s_k, bin = 255;
      for (; bin > 0; --bin) {
        int c = (int)hist[bin];
        if (kk <= c) break;
        kk -= c;
      }
      s_pref = pref | ((unsigned)bin << sh);
      s_k    = kk;
    }
    __syncthreads();
  }
  const unsigned Tu = s_pref;
  const int krem    = s_k;

  float smse = 0.0f, sind = 0.0f, szt = 0.0f;
  for (int i = t; i < Hh; i += 256) {
    unsigned u = su[i];
    bool keep = (u > Tu);
    if (u == Tu) {                      // deterministic index-order tie-break
      int r = 0;
      for (int j = 0; j < i; ++j) r += (su[j] == Tu);
      keep = (r < krem);
    }
    float z  = zr[i];
    float zl = Zl[(size_t)b * Hh + i];
    float zm = keep ? z : 0.0f;
    float d  = zm - zl;
    smse += d * d;
    sind += fabsf(d);
    szt  += fabsf(zm);
  }
  __shared__ float red[3][256];
  red[0][t] = smse; red[1][t] = sind; red[2][t] = szt;
  __syncthreads();
  for (int off = 128; off; off >>= 1) {
    if (t < off) {
      red[0][t] += red[0][t + off];
      red[1][t] += red[1][t + off];
      red[2][t] += red[2][t + off];
    }
    __syncthreads();
  }
  if (t == 0) {
    part3[b * 3 + 0] = red[0][0];
    part3[b * 3 + 1] = red[1][0];
    part3[b * 3 + 2] = red[2][0];
  }
}

// ---------------------------------------------------------------------------
// Grid-strided |tril(M,1)| reduction -> one partial per block.
// ---------------------------------------------------------------------------
__global__ __launch_bounds__(256)
void reduce_abs_tril_kernel(const float* __restrict__ Mx,
                            float* __restrict__ part)
{
  const long long n = (long long)Hh * Hh;
  long long i0 = (long long)blockIdx.x * 256 + threadIdx.x;
  long long st = (long long)gridDim.x * 256;
  float s = 0.0f;
  for (long long i = i0; i < n; i += st) {
    int h = (int)(i / Hh), d = (int)(i % Hh);
    if (d <= h + 1) s += fabsf(Mx[i]);
  }
  __shared__ float red[256];
  red[threadIdx.x] = s; __syncthreads();
  for (int off = 128; off; off >>= 1) {
    if ((int)threadIdx.x < off) red[threadIdx.x] += red[threadIdx.x + off];
    __syncthreads();
  }
  if (threadIdx.x == 0) part[blockIdx.x] = red[0];
}

// ---------------------------------------------------------------------------
// Single-block deterministic finalize -> six scalar losses in reference order.
// ---------------------------------------------------------------------------
__global__ __launch_bounds__(256)
void finalize_kernel(const float* __restrict__ part_bs,   // 512 (x8 multiplicity)
                     const float* __restrict__ part_m,    // 1024
                     const float* __restrict__ part_xt,   // 32
                     const float* __restrict__ part3,     // 512 x 3
                     float* __restrict__ out)
{
  __shared__ float red[256];
  const int t = threadIdx.x;
  auto blockSum = [&](float v) -> float {
    red[t] = v; __syncthreads();
    for (int off = 128; off; off >>= 1) {
      if (t < off) red[t] += red[t + off];
      __syncthreads();
    }
    float r = red[0]; __syncthreads();
    return r;
  };
  float s;
  s = 0.0f; for (int i = t; i < 512; i += 256) s += part_bs[i];
  float Sbs = blockSum(s) * (1.0f / 8.0f);        // undo M-tile multiplicity
  s = 0.0f; for (int i = t; i < 1024; i += 256) s += part_m[i];
  float Sm = blockSum(s);
  s = 0.0f; for (int i = t; i < 32; i += 256) s += part_xt[i];
  float Sxt = blockSum(s);
  float a = 0.0f, bI = 0.0f, c = 0.0f;
  for (int i = t; i < Bsz; i += 256) {
    a  += part3[i * 3 + 0];
    bI += part3[i * 3 + 1];
    c  += part3[i * 3 + 2];
  }
  float Smse = blockSum(a);
  float Sind = blockSum(bI);
  float Szt  = blockSum(c);
  if (t == 0) {
    out[0] = Sxt  / (float)((long long)Bsz * Dd);   // loss_mse_Xt
    out[1] = Smse / (float)((long long)Bsz * Hh);   // loss_mse_Zt
    out[2] = Sind / (float)((long long)Bsz * Hh);   // loss_indep
    out[3] = Sbs  / (float)((long long)Hh * Hh);    // loss_sparse_Bs
    out[4] = Sm   / (float)((long long)Hh * Hh);    // loss_sparse_M
    out[5] = Szt  / (float)((long long)Bsz * Hh);   // loss_sparse_Zt
  }
}

// ---------------------------------------------------------------------------
extern "C" void kernel_launch(void* const* d_in, const int* in_sizes, int n_in,
                              void* d_out, int out_size, void* d_ws, size_t ws_size,
                              hipStream_t stream)
{
  const float* Xp   = (const float*)d_in[0];   // (512, 1024, 21)
  const float* Fenc = (const float*)d_in[1];   // (1024, 4096)
  const float* Fdec = (const float*)d_in[2];   // (4096, 1024)
  const float* Bs   = (const float*)d_in[3];   // (20, 4096, 4096)
  const float* Mm   = (const float*)d_in[4];   // (4096, 4096)
  float* out = (float*)d_out;
  (void)Bs;

  char* ws = (char*)d_ws;
  size_t off = 0;
  unsigned short* Zp16 = (unsigned short*)(ws + off); off += (size_t)Tt * Bsz * Hh * 2;  // 88 MB
  float* Zlast   = (float*)(ws + off); off += (size_t)Bsz * Hh * 4;                      //  8 MB
  float* Zt      = (float*)(ws + off); off += (size_t)Bsz * Hh * 4;                      //  8 MB
  float* part_bs = (float*)(ws + off); off += 512 * 4;
  float* part_m  = (float*)(ws + off); off += 1024 * 4;
  float* part_xt = (float*)(ws + off); off += 64 * 4;
  float* part3   = (float*)(ws + off); off += (size_t)Bsz * 3 * 4;
  (void)ws_size; (void)in_sizes; (void)n_in; (void)out_size;

  // 1) Zp[t] = Xp[:,:,t] @ F_enc   (bf16 out; t==TAU also f32 Z_last)
  gemm_kernel<0, 128, 128, 2, 4>
      <<<dim3(Hh / 128, Bsz / 128, Tt), 256, 0, stream>>>(
          Xp, Fenc, Bs, Mm, Fdec, Zp16, Zlast, Zt, nullptr);

  // 2) Zt = sum_l Zp_rev[l] @ Bs[l]^T + Z_last @ tril(M,1)^T  (fixed order)
  //    + fused per-block partials of sum|Bs|
  gemm_kernel<1, 64, 64, 4, 2>
      <<<dim3(Hh / 64, Bsz / 64, 1), 256, 0, stream>>>(
          Xp, Fenc, Bs, Mm, Fdec, Zp16, Zlast, Zt, part_bs);

  // 3) fused mean((Z_last @ F_dec) - Xp[:,:,-1])^2 partials
  gemm_kernel<2, 128, 128, 2, 4>
      <<<dim3(Dd / 128, Bsz / 128, 1), 256, 0, stream>>>(
          Xp, Fenc, Bs, Mm, Fdec, Zp16, Zlast, Zt, part_xt);

  // 4) per-row top-100 mask + masked losses
  topk_kernel<<<dim3(Bsz), 256, 0, stream>>>(Zt, Zlast, part3);

  // 5) loss_sparse_M reduction (67 MB, minor)
  reduce_abs_tril_kernel<<<dim3(1024), 256, 0, stream>>>(Mm, part_m);

  // 6) deterministic finalize
  finalize_kernel<<<dim3(1), 256, 0, stream>>>(part_bs, part_m, part_xt, part3, out);
}
